// CoconAttention_23055384445521
// MI455X (gfx1250) — matrix operand order repaired and checked by reading, not compiled, not run on hardware
//
#include <hip/hip_runtime.h>

// ---------------------------------------------------------------------------
// CDNA5 (gfx1250) fused Cocon attention: bf16 WMMA GEMMs + flash attention.
// B=4, T=896, Tc=128, S=1024, NX=1024, H=16, Dh=64. wave32 only.
// GEMM: 128x128 block tile; A slab staged by TDM (tensor_load_to_lds),
// B slab staged by global_load_async_to_lds_b128; double-buffered LDS.
// ---------------------------------------------------------------------------

typedef __attribute__((ext_vector_type(16))) __bf16 v16bf;
typedef __attribute__((ext_vector_type(8)))  float  v8f;
typedef __attribute__((ext_vector_type(4)))  unsigned su4;
typedef __attribute__((ext_vector_type(8)))  unsigned su8;

struct __align__(16) U4 { unsigned a, b, c, d; };
union Frag { v16bf v; U4 q[2]; };

__device__ __forceinline__ unsigned short f2bf(float f) {
  unsigned u; __builtin_memcpy(&u, &f, 4);
  u += 0x7FFFu + ((u >> 16) & 1u);            // round-to-nearest-even
  return (unsigned short)(u >> 16);
}
__device__ __forceinline__ float bf2f(unsigned short h) {
  unsigned u = ((unsigned)h) << 16; float f; __builtin_memcpy(&f, &u, 4); return f;
}

__device__ __forceinline__ v8f wmma_bf16(const Frag& a, const Frag& b, v8f c) {
  // D = A(16x32 bf16) * B(32x16 bf16) + C(16x16 f32)
  return __builtin_amdgcn_wmma_f32_16x16x32_bf16(false, a.v, false, b.v,
                                                 (short)0, c, false, false);
}

// A-fragment (16x32 bf16). lane m = lane&15, half h = lane>>4.
// v0..3: K = h*8+0..7 ; v4..7: K = 16+h*8+0..7  (ISA 7.12.2)
__device__ __forceinline__ void load_fragA(Frag& f, const unsigned short* rowp, int hh) {
  f.q[0] = *(const U4*)(rowp + hh * 8);
  f.q[1] = *(const U4*)(rowp + 16 + hh * 8);
}
// B-fragment (32x16 bf16). lane n = lane&15, half h = lane>>4.
// v0..3: K = h*8+0..7 ; v4..7: K = 16+h*8+0..7 (same K order as A).
__device__ __forceinline__ void load_fragB(Frag& f, const unsigned short* colp, int hh) {
  f.q[0] = *(const U4*)(colp + hh * 8);
  f.q[1] = *(const U4*)(colp + 16 + hh * 8);
}

// ---- gfx1250 async global->LDS copy (ASYNCcnt-tracked, bypasses VGPRs) ----
__device__ __forceinline__ void async_b128(const unsigned short* g, unsigned short* l) {
  unsigned lds = (unsigned)(size_t)(void*)l;  // LDS aperture: low 32 bits = LDS offset
  asm volatile("global_load_async_to_lds_b128 %0, %1, off"
               :: "v"(lds), "v"(g) : "memory");
}
__device__ __forceinline__ void wait_async0() {
  asm volatile("s_wait_asynccnt 0x0" ::: "memory");
}
__device__ __forceinline__ void wait_async2() {
  asm volatile("s_wait_asynccnt 0x2" ::: "memory");
}

// ---- gfx1250 Tensor Data Mover: 2D tile (rows x rowBytes) -> LDS ----------
// D# per CDNA5 ISA. data_size = 8 bytes (code 3); all
// byte quantities must be multiples of 8. Tile lands contiguously in LDS.
__device__ __forceinline__ void tdm_load_2d(const void* gaddr, unsigned lds_off,
                                            unsigned rowBytes, unsigned rows,
                                            unsigned rowStrideBytes) {
  unsigned long long ga = (unsigned long long)(size_t)gaddr;
  unsigned td0 = rowBytes >> 3;                 // dim0 length / tile_dim0 (8B units)
  unsigned st0 = rowStrideBytes >> 3;           // tensor_dim0_stride (8B units)
  su4 g0;
  g0[0] = 0x1u;                                 // count=1, user descriptor
  g0[1] = lds_off;                              // lds_addr (bytes)
  g0[2] = (unsigned)ga;                         // global_addr[31:0]
  g0[3] = (unsigned)((ga >> 32) & 0x01FFFFFFu) | (2u << 30); // addr[56:32]|type=2
  su8 g1;
  g1[0] = (3u << 16);                           // data_size=8B; no multicast/pad
  g1[1] = (td0 & 0xFFFFu) << 16;                // tensor_dim0[15:0]
  g1[2] = ((td0 >> 16) & 0xFFFFu) | ((rows & 0xFFFFu) << 16); // td0 hi | td1 lo
  g1[3] = ((rows >> 16) & 0xFFFFu) | (td0 << 16);             // td1 hi | tile_dim0
  g1[4] = rows & 0xFFFFu;                       // tile_dim1 (tile_dim2 = 0)
  g1[5] = st0;                                  // tensor_dim0_stride[31:0]
  g1[6] = 0;                                    // stride0 hi | stride1 lo
  g1[7] = 0;                                    // stride1 hi
  asm volatile("tensor_load_to_lds %0, %1" :: "s"(g0), "s"(g1) : "memory");
}

// ---------------------------------------------------------------------------
// fp32 -> bf16 elementwise
// ---------------------------------------------------------------------------
__global__ __launch_bounds__(256) void conv_bf16(const float* __restrict__ in,
                                                 unsigned short* __restrict__ out, int n) {
  int i = blockIdx.x * 256 + threadIdx.x;
  if (i < n) out[i] = f2bf(in[i]);
}

// fp32 [K,N] -> bf16 [N,K] (transpose so WMMA B-fragments are K-contiguous)
__global__ __launch_bounds__(256) void convT_bf16(const float* __restrict__ in,
                                                  unsigned short* __restrict__ out,
                                                  int K, int N) {
  size_t i = (size_t)blockIdx.x * 256 + threadIdx.x;
  if (i >= (size_t)K * N) return;
  int k = (int)(i % K);
  size_t n = i / K;
  out[n * K + k] = f2bf(in[(size_t)k * N + n]);
}

// ---------------------------------------------------------------------------
// bf16 GEMM: C[M,N] = A[M,K] * Bt[N,K]^T + bias[N]
// Block tile 128x128, BK=32, 256 threads = 8 waves (4m x 2n), wave tile 32x64
// (2x4 WMMA). A slab via TDM (wave 0), B slab via async-to-LDS (all waves),
// double-buffered. out_f32=0 -> bf16 output, out_f32=1 -> f32 output.
// ---------------------------------------------------------------------------
#define BM 128
#define BN 128
#define BK 32

__global__ __launch_bounds__(256) void gemm_bf16(const unsigned short* __restrict__ A,
                                                 const unsigned short* __restrict__ Bt,
                                                 const float* __restrict__ bias,
                                                 void* __restrict__ Cout,
                                                 int M, int N, int K, int out_f32) {
  __shared__ __align__(16) unsigned short As[2][BM * BK];   // 2 x 8 KB
  __shared__ __align__(16) unsigned short Bs[2][BN * BK];   // 2 x 8 KB
  const int t    = threadIdx.x;
  const int lane = t & 31;
  const int wave = t >> 5;
  const int wm   = wave >> 1;            // 0..3  (32 rows each)
  const int wn   = wave & 1;             // 0..1  (64 cols each)
  const int bm0  = blockIdx.y * BM;
  const int bn0  = blockIdx.x * BN;
  const int r    = lane & 15;
  const int hh   = lane >> 4;

  v8f acc[2][4] = {};
  const int kIters = K / BK;

  // A slab: one TDM descriptor per K-step, issued by wave 0 only
  // (TENSORcnt is per-wave; barrier publishes the data to all waves).
  auto stageA = [&](int buf, int k0) {
    if (wave == 0)
      tdm_load_2d(A + (size_t)bm0 * K + k0,
                  (unsigned)(size_t)(void*)&As[buf][0],
                  BK * 2 /*rowBytes*/, BM /*rows*/, K * 2 /*rowStrideBytes*/);
  };
  // B slab: 2 async b128 ops per thread -> ASYNCcnt += 2 per wave.
  auto stageB = [&](int buf, int k0) {
#pragma unroll
    for (int ph = 0; ph < 2; ++ph) {
      int idx = ph * 256 + t;
      int row = idx >> 2, ch = idx & 3;
      async_b128(Bt + (size_t)(bn0 + row) * K + k0 + ch * 8,
                 &Bs[buf][row * BK + ch * 8]);
    }
  };

  stageA(0, 0);
  stageB(0, 0);
  for (int kt = 0; kt < kIters; ++kt) {
    const int cur = kt & 1;
    if (kt + 1 < kIters) {
      stageA(cur ^ 1, (kt + 1) * BK);  // prefetch next slab into other buffer
      stageB(cur ^ 1, (kt + 1) * BK);
      if (kt + 2 < kIters)             // L2 prefetch two slabs ahead
        __builtin_prefetch(Bt + (size_t)(bn0 + (t >> 2)) * K + (kt + 2) * BK, 0, 1);
      if (wave == 0) __builtin_amdgcn_s_wait_tensorcnt(1); // in-order: cur A landed
      wait_async2();                   // in-order: cur B landed
    } else {
      if (wave == 0) __builtin_amdgcn_s_wait_tensorcnt(0);
      wait_async0();
    }
    __syncthreads();

    Frag af[2], bfr[4];
#pragma unroll
    for (int i = 0; i < 2; ++i)
      load_fragA(af[i], &As[cur][(wm * 32 + i * 16 + r) * BK], hh);
#pragma unroll
    for (int j = 0; j < 4; ++j)
      load_fragB(bfr[j], &Bs[cur][(wn * 64 + j * 16 + r) * BK], hh);
#pragma unroll
    for (int i = 0; i < 2; ++i)
#pragma unroll
      for (int j = 0; j < 4; ++j)
        acc[i][j] = wmma_bf16(af[i], bfr[j], acc[i][j]);
    __syncthreads();                   // all reads of `cur` done before overwrite
  }

  // C/D layout: VGPR vr -> row vr + 8*hh, col = lane&15 (ISA 7.12.2)
#pragma unroll
  for (int i = 0; i < 2; ++i)
#pragma unroll
    for (int j = 0; j < 4; ++j) {
      int col = bn0 + wn * 64 + j * 16 + r;
      float bv = bias[col];
#pragma unroll
      for (int vr = 0; vr < 8; ++vr) {
        int row = bm0 + wm * 32 + i * 16 + vr + hh * 8;
        float val = acc[i][j][vr] + bv;
        if (out_f32) ((float*)Cout)[(size_t)row * N + col] = val;
        else ((unsigned short*)Cout)[(size_t)row * N + col] = f2bf(val);
      }
    }
}

// ---------------------------------------------------------------------------
// Scatter qkv[3584,3072] -> Q[bh,T,64] (x 1/8), K[bh,S,64] (rows Tc..), Vt[bh,64,S]
// ---------------------------------------------------------------------------
__global__ __launch_bounds__(256) void scatter_qkv(const unsigned short* __restrict__ qkv,
                                                   unsigned short* __restrict__ Q,
                                                   unsigned short* __restrict__ Kh,
                                                   unsigned short* __restrict__ Vt) {
  size_t idx = (size_t)blockIdx.x * 256 + threadIdx.x;
  if (idx >= (size_t)3584 * 3072) return;
  int c = (int)(idx % 3072);
  size_t bt = idx / 3072;
  int b = (int)(bt / 896), tt = (int)(bt % 896);
  unsigned short val = qkv[idx];
  int which = c >> 10, cc = c & 1023;
  int h = cc >> 6, d = cc & 63;
  size_t bh = (size_t)b * 16 + h;
  if (which == 0) {
    Q[(bh * 896 + tt) * 64 + d] = f2bf(bf2f(val) * 0.125f);   // 1/sqrt(64)
  } else if (which == 1) {
    Kh[(bh * 1024 + 128 + tt) * 64 + d] = val;
  } else {
    Vt[(bh * 64 + d) * 1024 + (128 + tt)] = val;
  }
}

// kvc[512,2048] -> K rows 0..127, Vt cols 0..127
__global__ __launch_bounds__(256) void scatter_ctx(const unsigned short* __restrict__ kvc,
                                                   unsigned short* __restrict__ Kh,
                                                   unsigned short* __restrict__ Vt) {
  size_t idx = (size_t)blockIdx.x * 256 + threadIdx.x;
  if (idx >= (size_t)512 * 2048) return;
  int c = (int)(idx % 2048);
  size_t row = idx / 2048;
  int b = (int)(row / 128), tc = (int)(row % 128);
  unsigned short val = kvc[idx];
  int which = c >> 10, cc = c & 1023;
  int h = cc >> 6, d = cc & 63;
  size_t bh = (size_t)b * 16 + h;
  if (which == 0) Kh[(bh * 1024 + tc) * 64 + d] = val;
  else            Vt[(bh * 64 + d) * 1024 + tc] = val;
}

// ---------------------------------------------------------------------------
// Flash attention: one wave per 16-row Q tile; stream S in 32-column chunks.
// Masks: causal j<=i+128, self j!=i (-> -1e4), ctx bias -2 for j<128.
// ---------------------------------------------------------------------------
__global__ __launch_bounds__(256) void attn_fwd(const unsigned short* __restrict__ Q,
                                                const unsigned short* __restrict__ Kh,
                                                const unsigned short* __restrict__ Vt,
                                                unsigned short* __restrict__ Aout) {
  __shared__ __align__(16) unsigned short plds[8][16 * 32];  // per-wave P staging
  const int t = threadIdx.x;
  const int lane = t & 31, wave = t >> 5;
  const int tile = blockIdx.x * 8 + wave;            // 3584 tiles = 64 bh * 56 mtiles
  if (tile >= 3584) return;
  const int mt = tile % 56, bh = tile / 56;
  const int b = bh >> 4, h_head = bh & 15;
  const int i_base = mt * 16;
  const int r = lane & 15, hh = lane >> 4;
  unsigned short* P = plds[wave];

  const unsigned short* qb = Q  + ((size_t)bh * 896 + i_base) * 64;
  const unsigned short* kb = Kh + (size_t)bh * 1024 * 64;
  const unsigned short* vb = Vt + (size_t)bh * 64 * 1024;

  Frag qa[2];                                        // Q tile 16x64 (two K-windows)
#pragma unroll
  for (int kw = 0; kw < 2; ++kw) load_fragA(qa[kw], qb + r * 64 + kw * 32, hh);

  v8f o[4] = {};
  float mstate[8], lstate[8];
#pragma unroll
  for (int i = 0; i < 8; ++i) { mstate[i] = -3.0e38f; lstate[i] = 0.0f; }

  int n_chunks = (i_base + 143) / 32 + 1;            // cover j <= i_base+15+128
  if (n_chunks > 32) n_chunks = 32;

  for (int sc = 0; sc < n_chunks; ++sc) {
    const int j0 = sc * 32;
    // ---- scores: S[16x32] = Qtile @ K^T (4 WMMAs) ----
    v8f s[2] = {};
#pragma unroll
    for (int ct = 0; ct < 2; ++ct) {
      const unsigned short* krow = kb + (size_t)(j0 + ct * 16 + r) * 64;
#pragma unroll
      for (int kw = 0; kw < 2; ++kw) {
        Frag kf; load_fragB(kf, krow + kw * 32, hh);
        s[ct] = wmma_bf16(qa[kw], kf, s[ct]);
      }
    }
    // ---- masks + online softmax ----
    float pm[2][8], cmax[8];
#pragma unroll
    for (int vr = 0; vr < 8; ++vr) {
      int i_g = i_base + vr + hh * 8;
      float best = -3.0e38f;
#pragma unroll
      for (int ct = 0; ct < 2; ++ct) {
        int j_g = j0 + ct * 16 + r;
        float v = s[ct][vr];
        bool allowed = (j_g <= i_g + 128) && (j_g != i_g);
        v = allowed ? v : -10000.0f;
        v += (j_g < 128) ? -2.0f : 0.0f;
        pm[ct][vr] = v;
        best = fmaxf(best, v);
      }
      cmax[vr] = best;
    }
#pragma unroll
    for (int m = 1; m < 16; m <<= 1)
#pragma unroll
      for (int vr = 0; vr < 8; ++vr)
        cmax[vr] = fmaxf(cmax[vr], __shfl_xor(cmax[vr], m, 32));

    float alpha[8], csum[8];
#pragma unroll
    for (int vr = 0; vr < 8; ++vr) {
      float mnew = fmaxf(mstate[vr], cmax[vr]);
      alpha[vr] = __expf(mstate[vr] - mnew);
      float p0 = __expf(pm[0][vr] - mnew);
      float p1 = __expf(pm[1][vr] - mnew);
      pm[0][vr] = p0; pm[1][vr] = p1;
      csum[vr] = p0 + p1;
      mstate[vr] = mnew;
    }
#pragma unroll
    for (int m = 1; m < 16; m <<= 1)
#pragma unroll
      for (int vr = 0; vr < 8; ++vr)
        csum[vr] += __shfl_xor(csum[vr], m, 32);
#pragma unroll
    for (int vr = 0; vr < 8; ++vr) lstate[vr] = lstate[vr] * alpha[vr] + csum[vr];
#pragma unroll
    for (int nt = 0; nt < 4; ++nt)
#pragma unroll
      for (int vr = 0; vr < 8; ++vr) o[nt][vr] *= alpha[vr];

    // ---- transpose P through LDS (C layout -> A-fragment layout) ----
#pragma unroll
    for (int vr = 0; vr < 8; ++vr) {
      int prow = vr + hh * 8;
      P[prow * 32 + r]      = f2bf(pm[0][vr]);
      P[prow * 32 + 16 + r] = f2bf(pm[1][vr]);
    }
    Frag pa; load_fragA(pa, P + r * 32, hh);   // same-wave DS ops are in-order

    // ---- O += P @ V (4 WMMAs over Dh) ----
#pragma unroll
    for (int nt = 0; nt < 4; ++nt) {
      Frag vf; load_fragB(vf, vb + (size_t)(nt * 16 + r) * 1024 + j0, hh);
      o[nt] = wmma_bf16(pa, vf, o[nt]);
    }
  }

  // ---- normalize and write merged-head output in bf16 ----
#pragma unroll
  for (int vr = 0; vr < 8; ++vr) {
    int i_g = i_base + vr + hh * 8;
    size_t bt = (size_t)b * 896 + i_g;
    float inv = 1.0f / lstate[vr];
#pragma unroll
    for (int nt = 0; nt < 4; ++nt) {
      int col = h_head * 64 + nt * 16 + r;
      Aout[bt * 1024 + col] = f2bf(o[nt][vr] * inv);
    }
  }
}

// ---------------------------------------------------------------------------
// Host launcher
// ---------------------------------------------------------------------------
extern "C" void kernel_launch(void* const* d_in, const int* in_sizes, int n_in,
                              void* d_out, int out_size, void* d_ws, size_t ws_size,
                              hipStream_t stream) {
  (void)in_sizes; (void)n_in; (void)out_size; (void)ws_size;
  const float* x      = (const float*)d_in[0];
  const float* ctx    = (const float*)d_in[1];
  const float* w_ref  = (const float*)d_in[2];
  const float* b_ref  = (const float*)d_in[3];
  const float* w_attn = (const float*)d_in[4];
  const float* b_attn = (const float*)d_in[5];
  const float* w_proj = (const float*)d_in[6];
  const float* b_proj = (const float*)d_in[7];

  char* ws = (char*)d_ws;
  size_t off = 0;
  auto take = [&](size_t elems) {
    unsigned short* p = (unsigned short*)(ws + off);
    off = (off + elems * 2 + 255) & ~(size_t)255;
    return p;
  };
  unsigned short* xb  = take((size_t)3584 * 1024);
  unsigned short* cb  = take((size_t)512 * 1024);
  unsigned short* waT = take((size_t)3072 * 1024);
  unsigned short* wrT = take((size_t)2048 * 1024);
  unsigned short* wpT = take((size_t)1024 * 1024);
  unsigned short* qkv = take((size_t)3584 * 3072);
  unsigned short* kvc = take((size_t)512 * 2048);
  unsigned short* Qb  = take((size_t)64 * 896 * 64);
  unsigned short* Kb  = take((size_t)64 * 1024 * 64);
  unsigned short* Vt  = take((size_t)64 * 1024 * 64);
  unsigned short* Ab  = take((size_t)3584 * 1024);

  dim3 blk(256);
  conv_bf16 <<<14336, blk, 0, stream>>>(x,   xb, 3584 * 1024);
  conv_bf16 <<<2048,  blk, 0, stream>>>(ctx, cb, 512 * 1024);
  convT_bf16<<<12288, blk, 0, stream>>>(w_attn, waT, 1024, 3072);
  convT_bf16<<<8192,  blk, 0, stream>>>(w_ref,  wrT, 1024, 2048);
  convT_bf16<<<4096,  blk, 0, stream>>>(w_proj, wpT, 1024, 1024);

  // qkv = x @ w_attn + b_attn          (M=3584,N=3072,K=1024) -> bf16
  gemm_bf16<<<dim3(3072 / BN, 3584 / BM), blk, 0, stream>>>(xb, waT, b_attn, qkv,
                                                            3584, 3072, 1024, 0);
  // kv_ctx = context @ w_ref + b_ref   (M=512,N=2048,K=1024) -> bf16
  gemm_bf16<<<dim3(2048 / BN, 512 / BM), blk, 0, stream>>>(cb, wrT, b_ref, kvc,
                                                           512, 2048, 1024, 0);

  scatter_qkv<<<43008, blk, 0, stream>>>(qkv, Qb, Kb, Vt);
  scatter_ctx<<<4096,  blk, 0, stream>>>(kvc, Kb, Vt);

  attn_fwd<<<448, blk, 0, stream>>>(Qb, Kb, Vt, Ab);

  // out = a @ w_proj + b_proj          (M=3584,N=1024,K=1024) -> f32 d_out
  gemm_bf16<<<dim3(1024 / BN, 3584 / BM), blk, 0, stream>>>(Ab, wpT, b_proj, d_out,
                                                            3584, 1024, 1024, 1);
}
